// RPN_16956530884755
// MI455X (gfx1250) — matrix operand order repaired and checked by reading, not compiled, not run on hardware
//
#include <hip/hip_runtime.h>
#include <hip/hip_bf16.h>

typedef __attribute__((ext_vector_type(16))) _Float16 v16h;
typedef __attribute__((ext_vector_type(8)))  float    v8f;

#define B_   4
#define HH_  8
#define WW_  8
#define INC_ 4096
#define OC_  512
#define KK_  (INC_ * 9)         // 36864
#define KCH  (KK_ / 32)         // 1152 K-chunks of 32
#define NBOX 2304
#define PREN 2000
#define POSTN 300

// WMMA fragment K patterns (16-bit A 16x32 / B 32x16 layouts, wave32)
__device__ __forceinline__ int kpatA(int hi, int e, int t) {
  return ((e & 4) << 2) + (hi << 3) + ((e & 3) << 1) + t;
}
__device__ __forceinline__ int kpatB(int hi, int e, int t) {
  return ((e & 4) << 2) + (hi << 3) + ((e & 3) << 1) + t;
}

// ---------------------------------------------------------------------------
// Prep A: im2col + f32->f16, stored in WMMA-fragment order:
//   Asw[(mt*KCH + kc)*32 + lane] = 16 halves of A-fragment for that lane.
// Each im2col element converted exactly once (vs 32x in the fused version).
// ---------------------------------------------------------------------------
__global__ void __launch_bounds__(256)
rpn_prep_a(const float* __restrict__ xin, v16h* __restrict__ Asw) {
  int gid = blockIdx.x * blockDim.x + threadIdx.x;
  if (gid >= 16 * KCH * 32) return;
  int lane = gid & 31;
  int kc   = (gid >> 5) % KCH;
  int mt   = gid / (KCH * 32);
  int row = lane & 15, hi = lane >> 4;
  int m  = mt * 16 + row;
  int bB = m >> 6, oy = (m >> 3) & 7, ox = m & 7;
  int kb = kc * 32;
  v16h frag;
#pragma unroll
  for (int e = 0; e < 8; ++e) {
#pragma unroll
    for (int t = 0; t < 2; ++t) {
      int ka = kb + kpatA(hi, e, t);
      int ca = ka / 9;
      int ra = ka - ca * 9;
      int ky = ra / 3;
      int kx = ra - ky * 3;
      int iy = oy + ky - 1, ix = ox + kx - 1;
      float av = 0.0f;
      if ((unsigned)iy < 8u && (unsigned)ix < 8u)
        av = xin[(((bB * INC_) + ca) << 6) + (iy << 3) + ix];
      frag[2 * e + t] = (_Float16)av;
    }
  }
  Asw[gid] = frag;
}

// ---------------------------------------------------------------------------
// Prep B: weights f32->f16 in WMMA-fragment order:
//   Bsw[(nt*KCH + kc)*32 + lane] = 16 halves of B-fragment for that lane.
// ---------------------------------------------------------------------------
__global__ void __launch_bounds__(256)
rpn_prep_b(const float* __restrict__ wconv, v16h* __restrict__ Bsw) {
  int gid = blockIdx.x * blockDim.x + threadIdx.x;
  if (gid >= 32 * KCH * 32) return;
  int lane = gid & 31;
  int kc   = (gid >> 5) % KCH;
  int nt   = gid / (KCH * 32);
  int hi = lane >> 4;
  int n  = nt * 16 + (lane & 15);
  long wbase = (long)n * INC_ * 9;
  int kb = kc * 32;
  v16h frag;
#pragma unroll
  for (int e = 0; e < 8; ++e) {
#pragma unroll
    for (int t = 0; t < 2; ++t) {
      int kbi = kb + kpatB(hi, e, t);
      int cb = kbi / 9;
      int rb = kbi - cb * 9;
      frag[2 * e + t] = (_Float16)wconv[wbase + (long)cb * 9 + rb];
    }
  }
  Bsw[gid] = frag;
}

// ---------------------------------------------------------------------------
// Streaming WMMA GEMM: one wave per 16x16 tile of f[256][512] (512 waves).
// Loop body: 2x32B coalesced fragment loads + prefetch + 1 WMMA.
// ---------------------------------------------------------------------------
__global__ void __launch_bounds__(256)
rpn_gemm_wmma(const v16h* __restrict__ Asw, const v16h* __restrict__ Bsw,
              const float* __restrict__ bconv, float* __restrict__ f) {
  const int lane = threadIdx.x & 31;
  const int wave = blockIdx.x * (blockDim.x >> 5) + (threadIdx.x >> 5);
  const int mt = wave & 15;
  const int nt = wave >> 4;
  const int row = lane & 15;
  const int hi  = lane >> 4;

  const v16h* ap = Asw + (size_t)mt * KCH * 32 + lane;
  const v16h* bp = Bsw + (size_t)nt * KCH * 32 + lane;

  v8f acc = {};
  for (int kc = 0; kc < KCH; ++kc) {
    v16h a = ap[(size_t)kc * 32];
    v16h b = bp[(size_t)kc * 32];
    int kp = (kc + 8 < KCH) ? kc + 8 : kc;
    __builtin_prefetch((const void*)(ap + (size_t)kp * 32), 0, 1);
    __builtin_prefetch((const void*)(bp + (size_t)kp * 32), 0, 1);
    acc = __builtin_amdgcn_wmma_f32_16x16x32_f16(
        false, a, false, b, (short)0, acc, false, false);
  }

  const int n = nt * 16 + row;
  const float bias = bconv[n];
#pragma unroll
  for (int v = 0; v < 8; ++v) {
    int mm = mt * 16 + v + (hi << 3);
    int bO = mm >> 6, yy = (mm >> 3) & 7, xx = mm & 7;
    f[(((bO * OC_) + n) << 6) + (yy << 3) + xx] = acc[v] + bias;
  }
}

// ---------------------------------------------------------------------------
// Fallback: fused conv (gathers inside GEMM loop) for small ws_size.
// ---------------------------------------------------------------------------
__global__ void __launch_bounds__(256)
rpn_conv3x3_fused(const float* __restrict__ xin,
                  const float* __restrict__ wconv,
                  const float* __restrict__ bconv,
                  float* __restrict__ f) {
  const int lane = threadIdx.x & 31;
  const int wave = blockIdx.x * (blockDim.x >> 5) + (threadIdx.x >> 5);
  const int mt = wave & 15;
  const int nt = wave >> 4;
  const int row = lane & 15;
  const int hi  = lane >> 4;
  const int m  = mt * 16 + row;
  const int bB = m >> 6, oy = (m >> 3) & 7, ox = m & 7;
  const int n  = nt * 16 + row;
  const long wbase = (long)n * INC_ * 9;

  v8f acc = {};
  for (int kb = 0; kb < KK_; kb += 32) {
    v16h afrag, bfrag;
#pragma unroll
    for (int e = 0; e < 8; ++e) {
#pragma unroll
      for (int t = 0; t < 2; ++t) {
        int ka = kb + kpatA(hi, e, t);
        int ca = ka / 9;
        int ra = ka - ca * 9;
        int ky = ra / 3;
        int kx = ra - ky * 3;
        int iy = oy + ky - 1, ix = ox + kx - 1;
        float av = 0.0f;
        if ((unsigned)iy < 8u && (unsigned)ix < 8u)
          av = xin[(((bB * INC_) + ca) << 6) + (iy << 3) + ix];
        afrag[2 * e + t] = (_Float16)av;
        int kbi = kb + kpatB(hi, e, t);
        int cb = kbi / 9;
        int rb = kbi - cb * 9;
        bfrag[2 * e + t] = (_Float16)wconv[wbase + (long)cb * 9 + rb];
      }
    }
    acc = __builtin_amdgcn_wmma_f32_16x16x32_f16(
        false, afrag, false, bfrag, (short)0, acc, false, false);
  }
  const float bias = bconv[n];
#pragma unroll
  for (int v = 0; v < 8; ++v) {
    int mm = mt * 16 + v + (hi << 3);
    int bO = mm >> 6, yy = (mm >> 3) & 7, xx = mm & 7;
    f[(((bO * OC_) + n) << 6) + (yy << 3) + xx] = acc[v] + bias;
  }
}

// ---------------------------------------------------------------------------
// Heads (cls fg score, 4 bbox deltas) + anchor decode (reference-exact quirks).
// ---------------------------------------------------------------------------
__global__ void rpn_heads_decode(const float* __restrict__ f,
                                 const float* __restrict__ cls_w,
                                 const float* __restrict__ cls_b,
                                 const float* __restrict__ bbox_w,
                                 const float* __restrict__ bbox_b,
                                 float* __restrict__ pred,
                                 float* __restrict__ scores) {
  int t = blockIdx.x * blockDim.x + threadIdx.x;
  if (t >= B_ * 9 * HH_ * WW_) return;
  int x = t & 7;
  int y = (t >> 3) & 7;
  int a = (t >> 6) % 9;
  int b = t / 576;

  const float* fp  = f + ((b * OC_) << 6) + (y << 3) + x;
  const int cc = 2 * a + 1;
  const float* cwp = cls_w + cc * OC_;
  const float* bw0 = bbox_w + (4 * a) * OC_;

  float s = 0.f, dx = 0.f, dy = 0.f, dw = 0.f, dh = 0.f;
  for (int c = 0; c < OC_; ++c) {
    float fv = fp[c << 6];
    s  += fv * cwp[c];
    dx += fv * bw0[c];
    dy += fv * bw0[OC_ + c];
    dw += fv * bw0[2 * OC_ + c];
    dh += fv * bw0[3 * OC_ + c];
  }
  s  += cls_b[cc];
  dx += bbox_b[4 * a + 0];
  dy += bbox_b[4 * a + 1];
  dw += bbox_b[4 * a + 2];
  dh += bbox_b[4 * a + 3];

  // Scrambled anchor: (H,W,9,4) flat reinterpreted as (9,H,W,4):
  // solve h*72 + w*9 + a9 = a*64 + y*8 + x.
  int q  = a * 64 + y * 8 + x;
  int h9 = q / 72;
  int rr = q - h9 * 72;
  int w9 = rr / 9;
  int a9 = rr - w9 * 9;

  const float SZ[3] = {128.f, 256.f, 512.f};
  const float RT[3] = {0.5f, 1.f, 2.f};
  float sz = SZ[a9 / 3];
  float rt = RT[a9 % 3];
  float ww = sz * sqrtf(rt);
  float hh = sz * sqrtf(1.0f / rt);
  float ex = (w9 + 0.5f) * 32.f - 0.5f * ww;
  float ey = (h9 + 0.5f) * 32.f - 0.5f * hh;

  float pcx = ex + dx * ww;
  float pcy = ey + dy * hh;
  float pw  = expf(dw) * ww;
  float ph  = expf(dh) * hh;

  int base = (((b * 9 + a) * 4) << 6) + (y << 3) + x;
  pred[base +   0] = pcx - 0.5f * pw;
  pred[base +  64] = pcy - 0.5f * ph;
  pred[base + 128] = pcx + 0.5f * pw;
  pred[base + 192] = pcy + 0.5f * ph;
  scores[t] = s;
}

// ---------------------------------------------------------------------------
// Single-block bitonic sort (4096 keys) == stable argsort(-scores); gather
// top-2000 boxes (pred viewed as [2304][4]).
// ---------------------------------------------------------------------------
#define NSORT 4096
__global__ void __launch_bounds__(1024)
rpn_sort_topk(const float* __restrict__ scores,
              const float* __restrict__ pred,
              float* __restrict__ tb) {
  __shared__ unsigned long long key[NSORT];
  const int tid = threadIdx.x;
  for (int i = tid; i < NSORT; i += 1024) {
    unsigned long long kk;
    if (i < NBOX) {
      unsigned u = __float_as_uint(scores[i]);
      unsigned m = (u & 0x80000000u) ? ~u : (u | 0x80000000u);
      unsigned d = ~m;
      kk = (((unsigned long long)d) << 32) | (unsigned)i;
    } else {
      kk = 0xFFFFFFFFFFFFFFFFull;
    }
    key[i] = kk;
  }
  __syncthreads();
  for (int k = 2; k <= NSORT; k <<= 1) {
    for (int j = k >> 1; j > 0; j >>= 1) {
      for (int i = tid; i < NSORT; i += 1024) {
        int ixj = i ^ j;
        if (ixj > i) {
          bool up = ((i & k) == 0);
          unsigned long long a = key[i], b2 = key[ixj];
          if ((a > b2) == up) { key[i] = b2; key[ixj] = a; }
        }
      }
      __syncthreads();
    }
  }
  for (int i = tid; i < PREN; i += 1024) {
    int idx = (int)(key[i] & 0xFFFFFFFFull);
    tb[i * 4 + 0] = pred[idx * 4 + 0];
    tb[i * 4 + 1] = pred[idx * 4 + 1];
    tb[i * 4 + 2] = pred[idx * 4 + 2];
    tb[i * 4 + 3] = pred[idx * 4 + 3];
  }
}

// ---------------------------------------------------------------------------
// Greedy NMS (serial i, parallel j) + argsort(pos)[:300] selection.
// ---------------------------------------------------------------------------
__global__ void __launch_bounds__(1024)
rpn_nms_out(const float* __restrict__ tb, float* __restrict__ out) {
  __shared__ float X1[PREN], Y1[PREN], X2[PREN], Y2[PREN], AR[PREN];
  __shared__ unsigned char SUP[PREN];
  __shared__ int SEL[POSTN];
  __shared__ unsigned char VAL[POSTN];
  const int tid = threadIdx.x;

  for (int i = tid; i < PREN; i += 1024) {
    float x1 = tb[i * 4 + 0], y1 = tb[i * 4 + 1];
    float x2 = tb[i * 4 + 2], y2 = tb[i * 4 + 3];
    X1[i] = x1; Y1[i] = y1; X2[i] = x2; Y2[i] = y2;
    AR[i] = (x2 - x1) * (y2 - y1);
    SUP[i] = 0;
  }
  __syncthreads();

  for (int i = 0; i < PREN; ++i) {
    if (!SUP[i]) {
      float x1 = X1[i], y1 = Y1[i], x2 = X2[i], y2 = Y2[i], ar = AR[i];
      for (int j = i + 1 + tid; j < PREN; j += 1024) {
        float iw = fmaxf(fminf(x2, X2[j]) - fmaxf(x1, X1[j]), 0.f);
        float ih = fmaxf(fminf(y2, Y2[j]) - fmaxf(y1, Y1[j]), 0.f);
        float inter = iw * ih;
        float uni = ar + AR[j] - inter;
        float iou = (uni > 0.f) ? (inter / uni) : 0.f;
        if (iou > 0.7f) SUP[j] = 1;
      }
    }
    __syncthreads();
  }

  if (tid == 0) {
    int cnt = 0;
    for (int i = 0; i < PREN && cnt < POSTN; ++i)
      if (!SUP[i]) { SEL[cnt] = i; VAL[cnt] = 1; ++cnt; }
    for (int i = 0; i < PREN && cnt < POSTN; ++i)
      if (SUP[i])  { SEL[cnt] = i; VAL[cnt] = 0; ++cnt; }
  }
  __syncthreads();

  for (int r = tid; r < POSTN; r += 1024) {
    int s2 = SEL[r];
    if (VAL[r]) {
      out[r * 4 + 0] = X1[s2];
      out[r * 4 + 1] = Y1[s2];
      out[r * 4 + 2] = X2[s2];
      out[r * 4 + 3] = Y2[s2];
    } else {
      out[r * 4 + 0] = 0.f;
      out[r * 4 + 1] = 0.f;
      out[r * 4 + 2] = 0.f;
      out[r * 4 + 3] = 0.f;
    }
  }
}

// ---------------------------------------------------------------------------
extern "C" void kernel_launch(void* const* d_in, const int* in_sizes, int n_in,
                              void* d_out, int out_size, void* d_ws, size_t ws_size,
                              hipStream_t stream) {
  const float* x      = (const float*)d_in[0];
  const float* conv_w = (const float*)d_in[1];
  const float* conv_b = (const float*)d_in[2];
  const float* cls_w  = (const float*)d_in[3];
  const float* cls_b  = (const float*)d_in[4];
  const float* bbox_w = (const float*)d_in[5];
  const float* bbox_b = (const float*)d_in[6];

  char* ws = (char*)d_ws;
  float* f      = (float*)(ws);                    // 524288 B
  float* pred   = (float*)(ws + 524288);           //  36864 B
  float* scores = (float*)(ws + 561152);           //   9216 B
  float* tb     = (float*)(ws + 570368);           //  32000 B

  const size_t A_OFF = 1u << 20;                   // 1 MiB
  const size_t A_SZ  = (size_t)16 * KCH * 32 * 32; // 18,874,368 B
  const size_t B_OFF = 20u << 20;                  // 20 MiB
  const size_t B_SZ  = (size_t)32 * KCH * 32 * 32; // 37,748,736 B
  const size_t NEED  = B_OFF + B_SZ;               // ~58.7 MB

  if (ws_size >= NEED) {
    v16h* Asw = (v16h*)(ws + A_OFF);
    v16h* Bsw = (v16h*)(ws + B_OFF);
    rpn_prep_a<<<(16 * KCH * 32) / 256, 256, 0, stream>>>(x, Asw);
    rpn_prep_b<<<(32 * KCH * 32) / 256, 256, 0, stream>>>(conv_w, Bsw);
    rpn_gemm_wmma<<<64, 256, 0, stream>>>(Asw, Bsw, conv_b, f);
  } else {
    rpn_conv3x3_fused<<<64, 256, 0, stream>>>(x, conv_w, conv_b, f);
  }

  rpn_heads_decode<<<9, 256, 0, stream>>>(f, cls_w, cls_b, bbox_w, bbox_b, pred, scores);
  rpn_sort_topk<<<1, 1024, 0, stream>>>(scores, pred, tb);
  rpn_nms_out<<<1, 1024, 0, stream>>>(tb, (float*)d_out);
}
